// MultiheadAttention_24661702214453
// MI455X (gfx1250) — matrix operand order repaired and checked by reading, not compiled, not run on hardware
//
#include <hip/hip_runtime.h>
#include <stdint.h>

// Problem constants (fixed by the reference harness).
#define BB 2
#define SS 2048
#define FD 1024
#define HH 16
#define DD 64
#define F3 3072
#define MM (BB * SS)   // 4096 rows total

typedef __attribute__((ext_vector_type(16))) __bf16         v16bf;
typedef __attribute__((ext_vector_type(8)))  float          v8f;
typedef __attribute__((ext_vector_type(8)))  unsigned short us8;   // 16 bytes
typedef __attribute__((ext_vector_type(4)))  unsigned int   ui4;   // 16 bytes

union Frag { v16bf v; us8 h[2]; };
union Acc  { v8f   v; float f[8]; };

__device__ __forceinline__ unsigned short f2bf(float f) {
  union { float f; unsigned u; } x; x.f = f;
  unsigned r = x.u + 0x7fffu + ((x.u >> 16) & 1u);   // round-to-nearest-even
  return (unsigned short)(r >> 16);
}

__device__ __forceinline__ v8f wmma_bf16(v16bf a, v16bf b, v8f c) {
  // D(f32 16x16) = A(bf16 16x32) * B(bf16 32x16) + C
  return __builtin_amdgcn_wmma_f32_16x16x32_bf16(
      /*neg_a=*/false, a, /*neg_b=*/false, b,
      /*c_mod=*/(short)0, c, /*reuse_a=*/false, /*reuse_b=*/false);
}

// A fragment (16x32 bf16): caller passes &A[row(lane)][k0]; lane<16 holds K
// elements {0..7, 16..23}, lane>=16 holds {8..15, 24..31} (ISA 7.12.2).
__device__ __forceinline__ Frag load_a(const unsigned short* rowptr, int hi) {
  Frag f;
  const us8* p = (const us8*)(rowptr + hi * 8);
  f.h[0] = p[0];        // K = k0 + 8*hi + [0..7]
  f.h[1] = p[2];        // K = k0 + 16 + 8*hi + [0..7]
  return f;
}

// B fragment (32x16 bf16, supplied as rows of B^T): caller passes
// &W[col(lane)][k0]; lane<16 holds K {0..7, 16..23}, lane>=16 holds {8..15, 24..31}.
__device__ __forceinline__ Frag load_b(const unsigned short* colptr, int hi) {
  Frag f;
  const us8* p = (const us8*)(colptr + hi * 8);
  f.h[0] = p[0];        // K = k0 + 8*hi + [0..7]
  f.h[1] = p[2];        // K = k0 + 16 + 8*hi + [0..7]
  return f;
}

// ---------------------------------------------------------------------------
// fp32 -> bf16 bulk convert (4 elems / thread)
// ---------------------------------------------------------------------------
__global__ void k_cvt(const float* __restrict__ in,
                      unsigned short* __restrict__ out, int n) {
  int i = (blockIdx.x * blockDim.x + threadIdx.x) * 4;
  if (i + 4 > n) return;
  float4 v = *reinterpret_cast<const float4*>(in + i);
  unsigned long long pk =
      (unsigned long long)f2bf(v.x) |
      ((unsigned long long)f2bf(v.y) << 16) |
      ((unsigned long long)f2bf(v.z) << 32) |
      ((unsigned long long)f2bf(v.w) << 48);
  *reinterpret_cast<unsigned long long*>(out + i) = pk;
}

// ---------------------------------------------------------------------------
// Software-pipelined 32(M) x 64(N) x FD(K) wave tile:
// 2 A-fragments + 4 shared B-fragments -> 8 WMMA per K-step, with next
// K-step's 12 b128 loads in flight while the current WMMAs execute.
// ---------------------------------------------------------------------------
__device__ __forceinline__ void gemm_core_32x64(
    const unsigned short* __restrict__ arow0,   // &A[(m0   +lr)][0]
    const unsigned short* __restrict__ arow1,   // &A[(m0+16+lr)][0]
    const unsigned short* const* bcol,          // 4x &W[(n0+t*16+lr)][0]
    int hi, Acc acc0[4], Acc acc1[4]) {
  Frag a0c = load_a(arow0, hi);
  Frag a1c = load_a(arow1, hi);
  Frag bc[4];
#pragma unroll
  for (int t = 0; t < 4; ++t) bc[t] = load_b(bcol[t], hi);

  for (int k0 = 0; k0 < FD; k0 += 32) {
    const int kn = (k0 + 32 < FD) ? (k0 + 32) : 0;   // wrap: result unused
    Frag a0n = load_a(arow0 + kn, hi);
    Frag a1n = load_a(arow1 + kn, hi);
    Frag bn[4];
#pragma unroll
    for (int t = 0; t < 4; ++t) bn[t] = load_b(bcol[t] + kn, hi);
    __builtin_prefetch(arow0 + k0 + 256, 0, 1);      // global_prefetch_b8

#pragma unroll
    for (int t = 0; t < 4; ++t) {
      acc0[t].v = wmma_bf16(a0c.v, bc[t].v, acc0[t].v);
      acc1[t].v = wmma_bf16(a1c.v, bc[t].v, acc1[t].v);
    }
    a0c = a0n;
    a1c = a1n;
#pragma unroll
    for (int t = 0; t < 4; ++t) bc[t] = bn[t];
  }
}

// ---------------------------------------------------------------------------
// QKV projection: C[M=4096, N=3072] = Xb @ Wqkv^T + bias
// Epilogue scatters into Q[B,H,S,D], K[B,H,S,D], Vt[B,H,D,S] (V transposed
// so the later P·V WMMA B-operand is load-contiguous).
// ---------------------------------------------------------------------------
__device__ __forceinline__ void store_qkv_half(
    const Acc acc[4], int m0, int n0, const float* __restrict__ bias,
    unsigned short* __restrict__ Qb, unsigned short* __restrict__ Kb,
    unsigned short* __restrict__ Vtb, int hi, int lr) {
  const int region = n0 / FD;        // 0=Q 1=K 2=V (64-wide tiles never span)
  const int nin    = n0 % FD;
  const int head   = nin / DD;       // tiles are head-aligned (DD==64)
  const int batch  = m0 / SS;
  const int sbase  = m0 % SS;

#pragma unroll
  for (int t = 0; t < 4; ++t) {
    const int   d  = t * 16 + lr;
    const float bv = bias[n0 + t * 16 + lr];
    if (region == 2) {
      // Vt[b][h][d][s]: lane holds fixed d, 8 consecutive s -> one b128 store.
      unsigned short* dst =
          Vtb + ((size_t)(batch * HH + head) * DD + d) * SS + sbase + hi * 8;
      ui4 pk;
#pragma unroll
      for (int j = 0; j < 4; ++j) {
        unsigned lo = f2bf(acc[t].f[2 * j] + bv);
        unsigned hb = f2bf(acc[t].f[2 * j + 1] + bv);
        pk[j] = lo | (hb << 16);
      }
      *reinterpret_cast<ui4*>(dst) = pk;
    } else {
      unsigned short* dst =
          (region == 0 ? Qb : Kb) +
          ((size_t)(batch * HH + head) * SS) * DD + d;
#pragma unroll
      for (int e = 0; e < 8; ++e) {
        int s = sbase + e + hi * 8;
        dst[(size_t)s * DD] = f2bf(acc[t].f[e] + bv);
      }
    }
  }
}

__global__ __launch_bounds__(128) void k_gemm_qkv(
    const unsigned short* __restrict__ xb,    // [MM][FD]
    const unsigned short* __restrict__ wb,    // [F3][FD]
    const float* __restrict__ bias,           // [F3]
    unsigned short* __restrict__ Qb,
    unsigned short* __restrict__ Kb,
    unsigned short* __restrict__ Vtb) {
  const int lane = threadIdx.x & 31;
  const int wave = threadIdx.x >> 5;
  const int hi   = lane >> 4;
  const int lr   = lane & 15;
  const int m0   = blockIdx.x * 128 + wave * 32;
  const int n0   = blockIdx.y * 64;

  Acc acc0[4], acc1[4];
#pragma unroll
  for (int t = 0; t < 4; ++t)
#pragma unroll
    for (int e = 0; e < 8; ++e) { acc0[t].f[e] = 0.f; acc1[t].f[e] = 0.f; }

  const unsigned short* arow0 = xb + (size_t)(m0 + lr) * FD;
  const unsigned short* arow1 = xb + (size_t)(m0 + 16 + lr) * FD;
  const unsigned short* bcol[4];
#pragma unroll
  for (int t = 0; t < 4; ++t)
    bcol[t] = wb + (size_t)(n0 + t * 16 + lr) * FD;

  gemm_core_32x64(arow0, arow1, bcol, hi, acc0, acc1);

  store_qkv_half(acc0, m0, n0, bias, Qb, Kb, Vtb, hi, lr);
  store_qkv_half(acc1, m0 + 16, n0, bias, Qb, Kb, Vtb, hi, lr);
}

// ---------------------------------------------------------------------------
// Causal flash attention. Grid (B*H, S/128); 8 waves/WG; wave owns 16 q-rows.
// Streams 32-key blocks: V^T fragments issued first (independent, overlap
// QK^T + softmax), S=QK^T (4 WMMA), online softmax (shfl_xor row reductions),
// P relayout C->A via per-wave LDS tile (+ s_wait_dscnt), O += P·V^T (4 WMMA).
// ---------------------------------------------------------------------------
__global__ __launch_bounds__(256) void k_attn(
    const unsigned short* __restrict__ Qb,
    const unsigned short* __restrict__ Kb,
    const unsigned short* __restrict__ Vtb,
    const unsigned char* __restrict__ pad,    // [BB][SS] (true = masked)
    unsigned short* __restrict__ attnb) {     // [BB][SS][FD]
  __shared__ unsigned short lds[8 * 16 * 32];

  const int lane = threadIdx.x & 31;
  const int wave = threadIdx.x >> 5;
  const int hi   = lane >> 4;
  const int lr   = lane & 15;
  const int bh   = blockIdx.x;
  const int b    = bh / HH, h = bh % HH;
  const int qw   = blockIdx.y * 128 + wave * 16;

  const unsigned short* Qh = Qb + (size_t)(b * HH + h) * SS * DD;
  const unsigned short* Kh = Kb + (size_t)(b * HH + h) * SS * DD;
  const unsigned short* Vh = Vtb + (size_t)(b * HH + h) * DD * SS;
  const unsigned char*  pm = pad + (size_t)b * SS;
  unsigned short*       pl = lds + wave * (16 * 32);

  Frag qf[2];
#pragma unroll
  for (int c = 0; c < 2; ++c)
    qf[c] = load_a(Qh + (size_t)(qw + lr) * DD + c * 32, hi);

  float mrow[8], lrow[8];
#pragma unroll
  for (int r = 0; r < 8; ++r) { mrow[r] = -3.0e38f; lrow[r] = 0.f; }
  Acc o[4];
#pragma unroll
  for (int t = 0; t < 4; ++t)
#pragma unroll
    for (int e = 0; e < 8; ++e) o[t].f[e] = 0.f;

  const float scale = 0.125f;   // 1/sqrt(64)

  for (int kj = 0; kj <= qw + 15; kj += 32) {
    // V^T fragments first: independent of softmax -> loads overlap everything
    // up to the P·V WMMAs at the bottom of the loop.
    Frag vb[4];
#pragma unroll
    for (int t = 0; t < 4; ++t)
      vb[t] = load_b(Vh + (size_t)(t * 16 + lr) * SS + kj, hi);

    Acc s0, s1;
#pragma unroll
    for (int e = 0; e < 8; ++e) { s0.f[e] = 0.f; s1.f[e] = 0.f; }

    // S tile 16x32: two 16x16 C-frags, K-dim (=D) accumulated over 2 chunks.
#pragma unroll
    for (int c = 0; c < 2; ++c) {
      Frag k0f = load_b(Kh + (size_t)(kj + lr) * DD + c * 32, hi);
      s0.v = wmma_bf16(qf[c].v, k0f.v, s0.v);
      Frag k1f = load_b(Kh + (size_t)(kj + 16 + lr) * DD + c * 32, hi);
      s1.v = wmma_bf16(qf[c].v, k1f.v, s1.v);
    }
    // Warm L2/L0 for the next key block while we do softmax VALU work.
    __builtin_prefetch(Kh + (size_t)(kj + 32 + lr) * DD, 0, 1);

    const int  kc0  = kj + lr, kc1 = kj + 16 + lr;
    const bool pad0 = pm[kc0] != 0;
    const bool pad1 = pm[kc1] != 0;

    float p0[8], p1[8], alpha[8];
#pragma unroll
    for (int r = 0; r < 8; ++r) {
      const int  q  = qw + r + hi * 8;
      const bool m0 = pad0 || (kc0 > q);
      const bool m1 = pad1 || (kc1 > q);
      float v0 = m0 ? -3.0e38f : s0.f[r] * scale;
      float v1 = m1 ? -3.0e38f : s1.f[r] * scale;
      float x  = fmaxf(v0, v1);
#pragma unroll
      for (int ms = 1; ms < 16; ms <<= 1)
        x = fmaxf(x, __shfl_xor(x, ms, 32));    // row-max over 16 lanes
      float mn = fmaxf(mrow[r], x);
      float a  = __expf(mrow[r] - mn);          // finite-sentinel: no NaN
      float e0 = m0 ? 0.f : __expf(v0 - mn);
      float e1 = m1 ? 0.f : __expf(v1 - mn);
      float sm = e0 + e1;
#pragma unroll
      for (int ms = 1; ms < 16; ms <<= 1)
        sm += __shfl_xor(sm, ms, 32);           // row-sum over 16 lanes
      lrow[r] = lrow[r] * a + sm;
      mrow[r] = mn;
      alpha[r] = a;
      p0[r] = e0;
      p1[r] = e1;
    }

    // Rescale running O by alpha (row r lives in element r of each frag).
#pragma unroll
    for (int t = 0; t < 4; ++t)
#pragma unroll
      for (int e = 0; e < 8; ++e) o[t].f[e] *= alpha[e];

    // P: C-layout -> LDS [16 rows][32 cols] bf16 -> reload in A-layout.
#pragma unroll
    for (int r = 0; r < 8; ++r) {
      pl[(r + hi * 8) * 32 + lr]      = f2bf(p0[r]);
      pl[(r + hi * 8) * 32 + 16 + lr] = f2bf(p1[r]);
    }
    asm volatile("s_wait_dscnt 0" ::: "memory");
    Frag pa = load_a(pl + lr * 32, hi);

    // O += P (16x32) * V^T-slice (32x16 per d-chunk); vb loaded at loop top.
#pragma unroll
    for (int t = 0; t < 4; ++t)
      o[t].v = wmma_bf16(pa.v, vb[t].v, o[t].v);
  }

  float inv[8];
#pragma unroll
  for (int r = 0; r < 8; ++r) inv[r] = 1.f / fmaxf(lrow[r], 1e-30f);

  unsigned short* obase = attnb + (size_t)b * SS * FD + (size_t)h * DD;
#pragma unroll
  for (int t = 0; t < 4; ++t) {
    const int d = t * 16 + lr;
#pragma unroll
    for (int e = 0; e < 8; ++e) {
      int s = qw + e + hi * 8;
      obase[(size_t)s * FD + d] = f2bf(o[t].f[e] * inv[e]);
    }
  }
}

// ---------------------------------------------------------------------------
// Output projection: out[M,F] = attn_bf16 @ Wo^T + out_b  (fp32 result)
// ---------------------------------------------------------------------------
__global__ __launch_bounds__(128) void k_gemm_out(
    const unsigned short* __restrict__ ab,    // [MM][FD]
    const unsigned short* __restrict__ wb,    // [FD][FD]
    const float* __restrict__ bias,           // [FD]
    float* __restrict__ out) {                // [MM][FD]
  const int lane = threadIdx.x & 31;
  const int wave = threadIdx.x >> 5;
  const int hi   = lane >> 4;
  const int lr   = lane & 15;
  const int m0   = blockIdx.x * 128 + wave * 32;
  const int n0   = blockIdx.y * 64;

  Acc acc0[4], acc1[4];
#pragma unroll
  for (int t = 0; t < 4; ++t)
#pragma unroll
    for (int e = 0; e < 8; ++e) { acc0[t].f[e] = 0.f; acc1[t].f[e] = 0.f; }

  const unsigned short* arow0 = ab + (size_t)(m0 + lr) * FD;
  const unsigned short* arow1 = ab + (size_t)(m0 + 16 + lr) * FD;
  const unsigned short* bcol[4];
#pragma unroll
  for (int t = 0; t < 4; ++t)
    bcol[t] = wb + (size_t)(n0 + t * 16 + lr) * FD;

  gemm_core_32x64(arow0, arow1, bcol, hi, acc0, acc1);

#pragma unroll
  for (int t = 0; t < 4; ++t) {
    const int   n  = n0 + t * 16 + lr;
    const float bv = bias[n];
#pragma unroll
    for (int e = 0; e < 8; ++e) {
      out[(size_t)(m0 + e + hi * 8) * FD + n]      = acc0[t].f[e] + bv;
      out[(size_t)(m0 + 16 + e + hi * 8) * FD + n] = acc1[t].f[e] + bv;
    }
  }
}

// ---------------------------------------------------------------------------
// Launch. Workspace layout (bf16 elements, ~48 MB total; fits in the 192 MB
// L2 so the multi-pass structure costs almost no HBM traffic):
//   xb[M*F] | wqkv[3F*F] | wo[F*F] | Q[B,H,S,D] | K[B,H,S,D] | Vt[B,H,D,S]
//   | attn[B,S,F]
// ---------------------------------------------------------------------------
extern "C" void kernel_launch(void* const* d_in, const int* in_sizes, int n_in,
                              void* d_out, int out_size, void* d_ws,
                              size_t ws_size, hipStream_t stream) {
  (void)in_sizes; (void)n_in; (void)out_size; (void)ws_size;

  const float*         x     = (const float*)d_in[0];
  const unsigned char* pad   = (const unsigned char*)d_in[1];
  const float*         qkv_w = (const float*)d_in[2];
  const float*         qkv_b = (const float*)d_in[3];
  const float*         out_w = (const float*)d_in[4];
  const float*         out_b = (const float*)d_in[5];
  float*               out   = (float*)d_out;

  unsigned short* xb    = (unsigned short*)d_ws;
  unsigned short* wqkvb = xb + (size_t)MM * FD;
  unsigned short* wob   = wqkvb + (size_t)F3 * FD;
  unsigned short* Qb    = wob + (size_t)FD * FD;
  unsigned short* Kb    = Qb + (size_t)BB * HH * SS * DD;
  unsigned short* Vtb   = Kb + (size_t)BB * HH * SS * DD;
  unsigned short* attnb = Vtb + (size_t)BB * HH * DD * SS;

  const int n1 = MM * FD, n2 = F3 * FD, n3 = FD * FD;
  k_cvt<<<n1 / 1024, 256, 0, stream>>>(x, xb, n1);
  k_cvt<<<n2 / 1024, 256, 0, stream>>>(qkv_w, wqkvb, n2);
  k_cvt<<<n3 / 1024, 256, 0, stream>>>(out_w, wob, n3);

  k_gemm_qkv<<<dim3(MM / 128, F3 / 64), 128, 0, stream>>>(
      xb, wqkvb, qkv_b, Qb, Kb, Vtb);
  k_attn<<<dim3(BB * HH, SS / 128), 256, 0, stream>>>(
      Qb, Kb, Vtb, pad, attnb);
  k_gemm_out<<<dim3(MM / 128, FD / 64), 128, 0, stream>>>(
      attnb, wob, out_b, out);
}
